// AttModule_17643725652463
// MI455X (gfx1250) — matrix operand order repaired and checked by reading, not compiled, not run on hardware
//
#include <hip/hip_runtime.h>
#include <hip/hip_bf16.h>

// ---------------------------------------------------------------------------
// Shapes
// ---------------------------------------------------------------------------
#define NB     128      // batch
#define SEQ    1024     // text length
#define INF_   768      // IN_F
#define TD     512      // TEXT_D
#define EMB_   512
#define NH     8
#define HD_    64
#define M1     1536     // N*12 rows of fc1
#define K2     6144     // 12*EMB fc2 reduction

typedef __attribute__((ext_vector_type(16))) __bf16 v16bf;
typedef __attribute__((ext_vector_type(8)))  float  v8f;

// ---------------------------------------------------------------------------
// CDNA5 async memory->LDS copy (ASYNCcnt-tracked, bypasses VGPRs).
// LDS byte address = low 32 bits of the generic shared pointer (flat LDS
// aperture truncates to addr[31:0]).
// ---------------------------------------------------------------------------
__device__ __forceinline__ void async_copy16(void* lds_dst, const void* gsrc) {
    unsigned int l = (unsigned int)(unsigned long long)lds_dst;
    asm volatile("global_load_async_to_lds_b128 %0, %1, off"
                 :: "v"(l), "v"(gsrc)
                 : "memory");
}
__device__ __forceinline__ void wait_async0() {
    asm volatile("s_wait_asynccnt 0" ::: "memory");
}

// ---------------------------------------------------------------------------
// fp32 -> bf16 conversion (one elem/thread; counts are multiples of 256)
// ---------------------------------------------------------------------------
__global__ __launch_bounds__(256)
void cvt_bf16(const float* __restrict__ src, __bf16* __restrict__ dst, int count) {
    int i = blockIdx.x * 256 + threadIdx.x;
    if (i < count) dst[i] = (__bf16)src[i];
}

// ---------------------------------------------------------------------------
// fc1: C[1536,512] = relu(A[1536,768] * W[512,768]^T + b)
// bf16 WMMA, block = 128 thr (4 waves), block tile 64(M) x 64(N), K step 32.
// Wave w owns rows [w*16, w*16+16); 4 accumulators along N.
// Output scattered into x[N, 12*EMB] layout (col = j*512 + e, row n=m/12).
// ---------------------------------------------------------------------------
__global__ __launch_bounds__(128)
void fc1_wmma(const __bf16* __restrict__ A, const __bf16* __restrict__ W,
              const float* __restrict__ bias, __bf16* __restrict__ xout) {
    __shared__ __align__(16) __bf16 sA[64 * 32];
    __shared__ __align__(16) __bf16 sB[64 * 32];
    const int tid = threadIdx.x;
    const int m0 = blockIdx.y * 64;
    const int n0 = blockIdx.x * 64;
    const int w = tid >> 5;
    const int lane = tid & 31;
    const int hs = lane >> 4;      // lane half (ISA 16-bit frag layout)
    const int l16 = lane & 15;

    v8f acc[4] = {};

    for (int k0 = 0; k0 < INF_; k0 += 32) {
        // Stage A & B tiles via async global->LDS (16B per lane per op)
        #pragma unroll
        for (int i = 0; i < 2; ++i) {
            int q = tid + i * 128;             // 256 chunks of 8 bf16
            int r = q >> 2, c8 = q & 3;
            async_copy16(&sA[r * 32 + c8 * 8],
                         A + (size_t)(m0 + r) * INF_ + k0 + c8 * 8);
            async_copy16(&sB[r * 32 + c8 * 8],
                         W + (size_t)(n0 + r) * INF_ + k0 + c8 * 8);
        }
        wait_async0();
        __syncthreads();

        // A fragment: lane half hs covers K {hs*8..hs*8+7} U {16+hs*8..}
        v16bf a;
        {
            const __bf16* base = &sA[(w * 16 + l16) * 32];
            #pragma unroll
            for (int i = 0; i < 8; ++i) {
                int kb = ((i >> 2) << 4) + (hs << 3) + ((i & 3) << 1);
                a[2 * i]     = base[kb];
                a[2 * i + 1] = base[kb + 1];
            }
        }
        #pragma unroll
        for (int c = 0; c < 4; ++c) {
            v16bf b;
            const __bf16* bb = &sB[(c * 16 + l16) * 32 + hs * 8];
            #pragma unroll
            for (int j = 0; j < 16; ++j) b[j] = bb[((j >> 3) << 4) + (j & 7)];
            acc[c] = __builtin_amdgcn_wmma_f32_16x16x32_bf16(
                false, a, false, b, (short)0, acc[c], false, false);
        }
        __syncthreads();
    }

    // Epilogue: bias + relu, scatter to [N, 12*512] bf16
    #pragma unroll
    for (int c = 0; c < 4; ++c) {
        int e = n0 + c * 16 + l16;
        float bv = bias[e];
        #pragma unroll
        for (int r = 0; r < 8; ++r) {
            int m = m0 + w * 16 + r + hs * 8;
            float val = acc[c][r] + bv;
            val = val > 0.f ? val : 0.f;
            int n = m / 12, j = m % 12;
            xout[(size_t)n * K2 + j * EMB_ + e] = (__bf16)val;
        }
    }
}

// ---------------------------------------------------------------------------
// fc2: vfeat[128,512] = relu(x[128,6144] * W2[512,6144]^T + b)  (fp32 out)
// block = 128 thr, tile 16(M) x 64(N); wave w owns one 16x16 col block.
// ---------------------------------------------------------------------------
__global__ __launch_bounds__(128)
void fc2_wmma(const __bf16* __restrict__ X, const __bf16* __restrict__ W,
              const float* __restrict__ bias, float* __restrict__ vfeat) {
    __shared__ __align__(16) __bf16 sA[16 * 32];
    __shared__ __align__(16) __bf16 sB[64 * 32];
    const int tid = threadIdx.x;
    const int m0 = blockIdx.y * 16;
    const int n0 = blockIdx.x * 64;
    const int w = tid >> 5;
    const int lane = tid & 31;
    const int hs = lane >> 4;
    const int l16 = lane & 15;

    v8f acc = {};

    for (int kk = 0; kk < K2; kk += 32) {
        if (tid < 64) {                         // A: 16x32 = 64 chunks
            int r = tid >> 2, c8 = tid & 3;
            async_copy16(&sA[r * 32 + c8 * 8],
                         X + (size_t)(m0 + r) * K2 + kk + c8 * 8);
        }
        #pragma unroll
        for (int i = 0; i < 2; ++i) {           // B: 64x32 = 256 chunks
            int q = tid + i * 128;
            int r = q >> 2, c8 = q & 3;
            async_copy16(&sB[r * 32 + c8 * 8],
                         W + (size_t)(n0 + r) * K2 + kk + c8 * 8);
        }
        wait_async0();
        __syncthreads();

        v16bf a;
        {
            const __bf16* base = &sA[l16 * 32];
            #pragma unroll
            for (int i = 0; i < 8; ++i) {
                int kb = ((i >> 2) << 4) + (hs << 3) + ((i & 3) << 1);
                a[2 * i]     = base[kb];
                a[2 * i + 1] = base[kb + 1];
            }
        }
        v16bf b;
        {
            const __bf16* bb = &sB[(w * 16 + l16) * 32 + hs * 8];
            #pragma unroll
            for (int j = 0; j < 16; ++j) b[j] = bb[((j >> 3) << 4) + (j & 7)];
        }
        acc = __builtin_amdgcn_wmma_f32_16x16x32_bf16(
            false, a, false, b, (short)0, acc, false, false);
        __syncthreads();
    }

    int e = n0 + w * 16 + l16;
    float bv = bias[e];
    #pragma unroll
    for (int r = 0; r < 8; ++r) {
        int m = m0 + r + hs * 8;
        float val = acc[r] + bv;
        vfeat[(size_t)m * EMB_ + e] = val > 0.f ? val : 0.f;
    }
}

// ---------------------------------------------------------------------------
// q projection: qs[n,e] = ((vfeat @ Wq^T)[n,e] + bq[e]) / 8
// ---------------------------------------------------------------------------
__global__ __launch_bounds__(256)
void qproj(const float* __restrict__ vfeat, const float* __restrict__ Wq,
           const float* __restrict__ bq, float* __restrict__ qs) {
    int o = blockIdx.x * 256 + threadIdx.x;   // 65536
    int n = o >> 9, e = o & 511;
    const float* a = vfeat + (size_t)n * EMB_;
    const float* w = Wq + (size_t)e * EMB_;
    float acc = bq[e];
    for (int k = 0; k < EMB_; k += 4)
        acc += a[k]*w[k] + a[k+1]*w[k+1] + a[k+2]*w[k+2] + a[k+3]*w[k+3];
    qs[o] = acc * 0.125f;
}

// ---------------------------------------------------------------------------
// Fold q into Wk:  qk[n,h,d] = sum_{d'} qs[n,h*64+d'] * Wk[h*64+d', d]
// (consecutive threads share q row, read consecutive d -> coalesced)
// ---------------------------------------------------------------------------
__global__ __launch_bounds__(256)
void qkfold(const float* __restrict__ qs, const float* __restrict__ Wk,
            float* __restrict__ qk) {
    int idx = blockIdx.x * 256 + threadIdx.x;  // 128*8*512
    int d = idx & 511, h = (idx >> 9) & 7, n = idx >> 12;
    const float* q = qs + (size_t)n * EMB_ + h * HD_;
    const float* w = Wk + (size_t)h * HD_ * TD + d;
    float acc = 0.f;
    #pragma unroll 8
    for (int dp = 0; dp < HD_; ++dp) acc += q[dp] * w[(size_t)dp * TD];
    qk[idx] = acc;
}

__global__ __launch_bounds__(256)
void qbk_kernel(const float* __restrict__ qs, const float* __restrict__ bk,
                float* __restrict__ qbk) {
    int idx = blockIdx.x * 256 + threadIdx.x;  // 1024
    if (idx >= NB * NH) return;
    int h = idx & 7, n = idx >> 3;
    const float* q = qs + (size_t)n * EMB_ + h * HD_;
    float acc = 0.f;
    for (int dp = 0; dp < HD_; ++dp) acc += q[dp] * bk[h * HD_ + dp];
    qbk[idx] = acc;
}

// ---------------------------------------------------------------------------
// Fused ragged attention, one workgroup per batch item (8 waves = 8 heads).
// Pass 1: scores[h,s] = qk[n,h,:] . t[n,s,:] + qbk   (t streamed in 8-row tiles
//         via async global->LDS)
// softmax (masked by t_len) done per-head by one wave with shfl reductions.
// Pass 2: u[h,:] = sum_s w[h,s] * t[n,s,:]  (t streamed again, L2-hot)
// then ctx[h,e'] = u[h,:] . Wv[h*64+e', :] + bv.
// ---------------------------------------------------------------------------
__global__ __launch_bounds__(256)
void attn_fused(const float* __restrict__ t, const int* __restrict__ t_len,
                const float* __restrict__ qk, const float* __restrict__ qbk,
                const float* __restrict__ Wv, const float* __restrict__ bv,
                float* __restrict__ ctx) {
    __shared__ float s_sc[NH * SEQ];     // 32 KB scores / weights / (reused) u
    __shared__ __align__(16) float s_t[8 * TD];   // 16 KB t tile
    __shared__ float s_red[256];         // partial dot reduction
    const int n = blockIdx.x;
    const int tid = threadIdx.x;
    const float* tn = t + (size_t)n * SEQ * TD;

    // ---- Pass 1: scores -------------------------------------------------
    const int h8 = tid & 7;              // head
    const int si = (tid >> 3) & 7;       // row in tile
    const int part = tid >> 6;           // k-quarter
    const float* qkh = qk + ((size_t)n * NH + h8) * TD;
    const float  qb  = qbk[n * NH + h8];

    for (int s0 = 0; s0 < SEQ; s0 += 8) {
        #pragma unroll
        for (int i = 0; i < 4; ++i) {
            int q = tid + i * 256;       // 1024 16B chunks
            async_copy16(&s_t[q * 4], tn + (size_t)s0 * TD + q * 4);
        }
        if (s0 + 8 < SEQ)                 // global_prefetch_b8 of next tile
            __builtin_prefetch(tn + (size_t)(s0 + 8) * TD + tid * 16, 0, 1);
        wait_async0();
        __syncthreads();

        float acc = 0.f;
        const float* tr = &s_t[si * TD + part * 128];
        const float* qp = qkh + part * 128;
        #pragma unroll 4
        for (int k = 0; k < 128; ++k) acc += qp[k] * tr[k];
        s_red[tid] = acc;
        __syncthreads();
        if (tid < 64) {
            float sc = s_red[tid] + s_red[tid + 64] +
                       s_red[tid + 128] + s_red[tid + 192] + qb;
            s_sc[h8 * SEQ + s0 + si] = sc;
        }
        // next iteration's post-load __syncthreads protects s_red / s_sc
    }
    __syncthreads();

    // ---- masked softmax: wave h handles head h --------------------------
    const int len = t_len[n];
    const int h = tid >> 5, l = tid & 31;
    float* sh = &s_sc[h * SEQ];
    float mx = -3.0e38f;
    for (int s = l; s < SEQ; s += 32) {
        float v = (s < len) ? sh[s] : -3.0e38f;
        sh[s] = v;
        mx = fmaxf(mx, v);
    }
    #pragma unroll
    for (int off = 16; off > 0; off >>= 1) mx = fmaxf(mx, __shfl_xor(mx, off, 32));
    float sum = 0.f;
    for (int s = l; s < SEQ; s += 32) {
        float p = __expf(sh[s] - mx);
        sh[s] = p;
        sum += p;
    }
    #pragma unroll
    for (int off = 16; off > 0; off >>= 1) sum += __shfl_xor(sum, off, 32);
    float inv = 1.f / sum;
    for (int s = l; s < SEQ; s += 32) sh[s] *= inv;
    __syncthreads();

    // ---- Pass 2: u[h,:] accumulation ------------------------------------
    float u[16];
    #pragma unroll
    for (int i = 0; i < 16; ++i) u[i] = 0.f;

    for (int s0 = 0; s0 < SEQ; s0 += 8) {
        #pragma unroll
        for (int i = 0; i < 4; ++i) {
            int q = tid + i * 256;
            async_copy16(&s_t[q * 4], tn + (size_t)s0 * TD + q * 4);
        }
        if (s0 + 8 < SEQ)
            __builtin_prefetch(tn + (size_t)(s0 + 8) * TD + tid * 16, 0, 1);
        wait_async0();
        __syncthreads();
        #pragma unroll
        for (int j = 0; j < 8; ++j) {
            float wgt = sh[s0 + j];              // LDS broadcast within wave h
            const float* trow = &s_t[j * TD + l];
            #pragma unroll
            for (int i = 0; i < 16; ++i) u[i] = fmaf(wgt, trow[i * 32], u[i]);
        }
        __syncthreads();
    }

    // stash u (weights dead now; reuse first 16 KB of s_sc)
    float* su = &s_sc[h * TD];
    #pragma unroll
    for (int i = 0; i < 16; ++i) su[l + i * 32] = u[i];
    __syncthreads();

    // ctx[h,e'] = u[h,:] . Wv[h*64+e', :] + bv
    for (int o = tid; o < EMB_; o += 256) {
        int hh = o >> 6, ee = o & 63;
        const float* wrow = Wv + (size_t)(hh * HD_ + ee) * TD;
        const float* uu = &s_sc[hh * TD];
        float acc = bv[hh * HD_ + ee];
        for (int k = 0; k < TD; k += 4)
            acc += uu[k]*wrow[k] + uu[k+1]*wrow[k+1] +
                   uu[k+2]*wrow[k+2] + uu[k+3]*wrow[k+3];
        ctx[(size_t)n * EMB_ + o] = acc;
    }
}

// ---------------------------------------------------------------------------
// Output projection: attn_out = ctx @ Wo^T + bo
// ---------------------------------------------------------------------------
__global__ __launch_bounds__(256)
void outproj(const float* __restrict__ ctx, const float* __restrict__ Wo,
             const float* __restrict__ bo, float* __restrict__ out) {
    int o = blockIdx.x * 256 + threadIdx.x;   // 65536
    int n = o >> 9, e = o & 511;
    const float* a = ctx + (size_t)n * EMB_;
    const float* w = Wo + (size_t)e * EMB_;
    float acc = bo[e];
    for (int k = 0; k < EMB_; k += 4)
        acc += a[k]*w[k] + a[k+1]*w[k+1] + a[k+2]*w[k+2] + a[k+3]*w[k+3];
    out[o] = acc;
}

// ---------------------------------------------------------------------------
// launch
// ---------------------------------------------------------------------------
extern "C" void kernel_launch(void* const* d_in, const int* in_sizes, int n_in,
                              void* d_out, int out_size, void* d_ws, size_t ws_size,
                              hipStream_t stream) {
    const float* v     = (const float*)d_in[0];
    const float* t     = (const float*)d_in[1];
    const int*   t_len = (const int*)  d_in[2];
    const float* W_fc1 = (const float*)d_in[3];
    const float* b_fc1 = (const float*)d_in[4];
    const float* W_fc2 = (const float*)d_in[5];
    const float* b_fc2 = (const float*)d_in[6];
    const float* Wq    = (const float*)d_in[7];
    const float* Wk    = (const float*)d_in[8];
    const float* Wv    = (const float*)d_in[9];
    const float* bq    = (const float*)d_in[10];
    const float* bk    = (const float*)d_in[11];
    const float* bv    = (const float*)d_in[12];
    const float* Wo    = (const float*)d_in[13];
    const float* bo    = (const float*)d_in[14];
    float* out = (float*)d_out;               // [0:65536) vfeat, [65536:) attn_out

    // workspace layout (all 16B-aligned)
    __bf16* v_bf    = (__bf16*)d_ws;                      // 1536*768
    __bf16* wfc1_bf = v_bf    + (size_t)M1 * INF_;        // 512*768
    __bf16* wfc2_bf = wfc1_bf + (size_t)EMB_ * INF_;      // 512*6144
    __bf16* x_bf    = wfc2_bf + (size_t)EMB_ * K2;        // 128*6144
    float*  qs      = (float*)(x_bf + (size_t)NB * K2);   // 128*512
    float*  qkbuf   = qs    + NB * EMB_;                  // 128*8*512
    float*  qbk     = qkbuf + NB * NH * TD;               // 128*8
    float*  ctx     = qbk   + NB * NH;                    // 128*512

    // bf16 conversions for WMMA operands
    cvt_bf16<<<(M1 * INF_) / 256, 256, 0, stream>>>(v, v_bf, M1 * INF_);
    cvt_bf16<<<(EMB_ * INF_) / 256, 256, 0, stream>>>(W_fc1, wfc1_bf, EMB_ * INF_);
    cvt_bf16<<<(EMB_ * K2) / 256, 256, 0, stream>>>(W_fc2, wfc2_bf, EMB_ * K2);

    // fc1: [1536,768] x [512,768]^T (WMMA), scattered to x[N, 12*512]
    fc1_wmma<<<dim3(EMB_ / 64, M1 / 64), 128, 0, stream>>>(v_bf, wfc1_bf, b_fc1, x_bf);
    // fc2: [128,6144] x [512,6144]^T (WMMA) -> vfeat (output #1)
    fc2_wmma<<<dim3(EMB_ / 64, NB / 16), 128, 0, stream>>>(x_bf, wfc2_bf, b_fc2, out);

    // q projection (+1/sqrt(HD)) and fold into Wk
    qproj<<<(NB * EMB_) / 256, 256, 0, stream>>>(out, Wq, bq, qs);
    qkfold<<<(NB * NH * TD) / 256, 256, 0, stream>>>(qs, Wk, qkbuf);
    qbk_kernel<<<(NB * NH + 255) / 256, 256, 0, stream>>>(qs, bk, qbk);

    // fused masked attention (streams t, never materializes K/V)
    attn_fused<<<NB, 256, 0, stream>>>(t, t_len, qkbuf, qbk, Wv, bv, ctx);

    // output projection (output #2)
    outproj<<<(NB * EMB_) / 256, 256, 0, stream>>>(ctx, Wo, bo, out + NB * EMB_);
}
